// ImprovedTransformerMMOT_43825846288823
// MI455X (gfx1250) — compile-verified, matrix operand order checked by reading, not run on hardware
//
#include <hip/hip_runtime.h>
#include <math.h>

// ---------------------------------------------------------------------------
// Problem constants (from reference)
// ---------------------------------------------------------------------------
#define BATCH 16
#define NP1   65
#define TOK   (BATCH * NP1)   // 1040 tokens
#define MGRID 256
#define DMODEL 512
#define NHEAD 8
#define DHEAD 64
#define NLAYER 3
#define DFFN  2048
#define EPS_PROJ 0.01f
#define LN_EPS 1e-5f

typedef _Float16 v16h __attribute__((ext_vector_type(16)));
typedef _Float16 v8h  __attribute__((ext_vector_type(8)));
typedef float    v8f  __attribute__((ext_vector_type(8)));

__device__ __forceinline__ int imin_(int a, int b) { return a < b ? a : b; }

// ---------------------------------------------------------------------------
// Fragment loader: 16x32 f16 tile (A-matrix layout; B uses the same pattern
// with "row" = N index since we compute A @ W^T with W stored N-major).
// CDNA5 16-bit A layout (ISA 7.12.2): lanes 0-15 hold row M=lane with
// K = 0..7 (VGPR0-3) and K = 16..23 (VGPR4-7); lanes 16-31 hold the same rows
// with K = 8..15 and K = 24..31.  => per lane: two contiguous 8-half chunks
// at kBase + 8*half and kBase + 8*half + 16.
// ---------------------------------------------------------------------------
__device__ __forceinline__ v16h load_frag16x32(const _Float16* __restrict__ base,
                                               int rowBase, int maxRow, int ld,
                                               int kBase) {
    int lane = threadIdx.x & 31;
    int half = lane >> 4;
    int r    = lane & 15;
    int row  = rowBase + r;
    row = row < maxRow ? row : (maxRow - 1);   // clamp (padded tiles)
    const _Float16* p = base + (size_t)row * (size_t)ld + (kBase + 8 * half);
    v8h lo = *(const v8h*)(p);
    v8h hi = *(const v8h*)(p + 16);
    return __builtin_shufflevector(lo, hi, 0,1,2,3,4,5,6,7,8,9,10,11,12,13,14,15);
}

// ---------------------------------------------------------------------------
// Generic WMMA GEMM:  out[M,N] = A[M,K](f16) @ W[N,K](f16)^T + bias
// act: 0 = none, 1 = exact GELU.  outF / outH optional (nullptr to skip).
// Block = 128 threads = 4 waves arranged 2x2 -> 64x64 block tile,
// each wave computes 32x32 via 2x2 wmma_f32_16x16x32_f16.
// ---------------------------------------------------------------------------
__global__ void __launch_bounds__(128)
gemm_wmma_f16(const _Float16* __restrict__ A, const _Float16* __restrict__ W,
              const float* __restrict__ bias, float* __restrict__ outF,
              _Float16* __restrict__ outH, int M, int N, int K, int act) {
    int tid  = threadIdx.x;
    int wave = tid >> 5;
    int lane = tid & 31;
    int mBase = blockIdx.y * 64 + (wave >> 1) * 32;
    int nBase = blockIdx.x * 64 + (wave & 1) * 32;

    v8f c00 = {}, c01 = {}, c10 = {}, c11 = {};

    for (int k = 0; k < K; k += 32) {
        if (k + 32 < K) {  // gfx1250 global_prefetch_b8 for the next K tile
            __builtin_prefetch(A + (size_t)imin_(mBase + lane, M - 1) * K + k + 32, 0, 1);
            __builtin_prefetch(W + (size_t)imin_(nBase + lane, N - 1) * K + k + 32, 0, 1);
        }
        v16h a0 = load_frag16x32(A, mBase,      M, K, k);
        v16h a1 = load_frag16x32(A, mBase + 16, M, K, k);
        v16h b0 = load_frag16x32(W, nBase,      N, K, k);
        v16h b1 = load_frag16x32(W, nBase + 16, N, K, k);
        c00 = __builtin_amdgcn_wmma_f32_16x16x32_f16(false, a0, false, b0, (short)0, c00, false, false);
        c01 = __builtin_amdgcn_wmma_f32_16x16x32_f16(false, a0, false, b1, (short)0, c01, false, false);
        c10 = __builtin_amdgcn_wmma_f32_16x16x32_f16(false, a1, false, b0, (short)0, c10, false, false);
        c11 = __builtin_amdgcn_wmma_f32_16x16x32_f16(false, a1, false, b1, (short)0, c11, false, false);
    }

    int half = lane >> 4;
    int r    = lane & 15;
    auto store_tile = [&](const v8f& c, int mT, int nT) {
        int n = nT + r;
        if (n >= N) return;
        float bv = bias ? bias[n] : 0.0f;
#pragma unroll
        for (int e = 0; e < 8; e++) {
            int m = mT + e + 8 * half;    // C/D layout: VGPR e -> M = e (+8 for hi lanes)
            if (m < M) {
                float v = c[e] + bv;
                if (act == 1) v = 0.5f * v * (1.0f + erff(v * 0.70710678118f));
                if (outF) outF[(size_t)m * N + n] = v;
                if (outH) outH[(size_t)m * N + n] = (_Float16)v;
            }
        }
    };
    store_tile(c00, mBase,      nBase);
    store_tile(c01, mBase,      nBase + 16);
    store_tile(c10, mBase + 16, nBase);
    store_tile(c11, mBase + 16, nBase + 16);
}

// ---------------------------------------------------------------------------
// Weight preprocessing: f32 -> f16 copy / transpose
// ---------------------------------------------------------------------------
__global__ void convert_f16_kernel(const float* __restrict__ src,
                                   _Float16* __restrict__ dst, long n) {
    long i = (long)blockIdx.x * blockDim.x + threadIdx.x;
    if (i < n) dst[i] = (_Float16)src[i];
}

// src is R x C (row-major); dst is C x R
__global__ void transpose_f16_kernel(const float* __restrict__ src,
                                     _Float16* __restrict__ dst, int R, int C) {
    int i = blockIdx.x * blockDim.x + threadIdx.x;
    if (i < R * C) {
        int rr = i / C, cc = i % C;
        dst[(size_t)cc * R + rr] = (_Float16)src[i];
    }
}

// ---------------------------------------------------------------------------
// Conv(k=5,pad=2) + mean over positions, algebraically reduced:
// emb[c] = conv_b[c] + (1/M) * sum_t conv_w[c,t] * S_t, with
// S_t = row_sum adjusted by 2 edge elements.
// One block per token (1040), 256 threads.
// ---------------------------------------------------------------------------
__global__ void __launch_bounds__(256)
embed_kernel(const float* __restrict__ marg, const float* __restrict__ convW,
             const float* __restrict__ convB, _Float16* __restrict__ emb) {
    __shared__ float red[256];
    __shared__ float S[5];
    int t = blockIdx.x;
    const float* row = marg + (size_t)t * MGRID;
    red[threadIdx.x] = row[threadIdx.x];
    __syncthreads();
    for (int s = 128; s > 0; s >>= 1) {
        if ((int)threadIdx.x < s) red[threadIdx.x] += red[threadIdx.x + s];
        __syncthreads();
    }
    if (threadIdx.x == 0) {
        float tot = red[0];
        float x0 = row[0], x1 = row[1], xa = row[MGRID - 2], xb = row[MGRID - 1];
        S[0] = tot - xa - xb;  // taps reaching past the right edge
        S[1] = tot - xb;
        S[2] = tot;
        S[3] = tot - x0;
        S[4] = tot - x0 - x1;
    }
    __syncthreads();
    if (threadIdx.x < 128) {
        int c = threadIdx.x;
        float acc = 0.0f;
#pragma unroll
        for (int tt = 0; tt < 5; tt++) acc += convW[c * 5 + tt] * S[tt];
        emb[(size_t)t * 128 + c] = (_Float16)(convB[c] + acc * (1.0f / MGRID));
    }
}

// ---------------------------------------------------------------------------
// xs = gelu(LN(pre)) + pos_enc ; write f32 master + f16 mirror.
// One block per token, 256 threads (2 elems of D=512 each).
// ---------------------------------------------------------------------------
__global__ void __launch_bounds__(256)
ln_gelu_pos_kernel(const float* __restrict__ pre, const float* __restrict__ g,
                   const float* __restrict__ b, const float* __restrict__ pos,
                   float* __restrict__ xsF, _Float16* __restrict__ xsH) {
    __shared__ float rs[256], rs2[256];
    int t = blockIdx.x;
    const float* pr = pre + (size_t)t * DMODEL;
    int i0 = threadIdx.x, i1 = threadIdx.x + 256;
    float v0 = pr[i0], v1 = pr[i1];
    rs[threadIdx.x]  = v0 + v1;
    rs2[threadIdx.x] = v0 * v0 + v1 * v1;
    __syncthreads();
    for (int s = 128; s > 0; s >>= 1) {
        if ((int)threadIdx.x < s) { rs[threadIdx.x] += rs[threadIdx.x + s];
                                    rs2[threadIdx.x] += rs2[threadIdx.x + s]; }
        __syncthreads();
    }
    float mu  = rs[0] * (1.0f / DMODEL);
    float var = rs2[0] * (1.0f / DMODEL) - mu * mu;
    float inv = rsqrtf(var + LN_EPS);
    int p = t % NP1;
    float y0 = (v0 - mu) * inv * g[i0] + b[i0];
    float y1 = (v1 - mu) * inv * g[i1] + b[i1];
    y0 = 0.5f * y0 * (1.0f + erff(y0 * 0.70710678118f)) + pos[(size_t)p * DMODEL + i0];
    y1 = 0.5f * y1 * (1.0f + erff(y1 * 0.70710678118f)) + pos[(size_t)p * DMODEL + i1];
    xsF[(size_t)t * DMODEL + i0] = y0;  xsH[(size_t)t * DMODEL + i0] = (_Float16)y0;
    xsF[(size_t)t * DMODEL + i1] = y1;  xsH[(size_t)t * DMODEL + i1] = (_Float16)y1;
}

// ---------------------------------------------------------------------------
// xs = LN(xs + delta) in place, refresh the f16 mirror.
// ---------------------------------------------------------------------------
__global__ void __launch_bounds__(256)
add_ln_kernel(float* __restrict__ xsF, const float* __restrict__ delta,
              const float* __restrict__ g, const float* __restrict__ b,
              _Float16* __restrict__ xsH) {
    __shared__ float rs[256], rs2[256];
    int t = blockIdx.x;
    float* xr = xsF + (size_t)t * DMODEL;
    const float* dr = delta + (size_t)t * DMODEL;
    int i0 = threadIdx.x, i1 = threadIdx.x + 256;
    float v0 = xr[i0] + dr[i0];
    float v1 = xr[i1] + dr[i1];
    rs[threadIdx.x]  = v0 + v1;
    rs2[threadIdx.x] = v0 * v0 + v1 * v1;
    __syncthreads();
    for (int s = 128; s > 0; s >>= 1) {
        if ((int)threadIdx.x < s) { rs[threadIdx.x] += rs[threadIdx.x + s];
                                    rs2[threadIdx.x] += rs2[threadIdx.x + s]; }
        __syncthreads();
    }
    float mu  = rs[0] * (1.0f / DMODEL);
    float var = rs2[0] * (1.0f / DMODEL) - mu * mu;
    float inv = rsqrtf(var + LN_EPS);
    float y0 = (v0 - mu) * inv * g[i0] + b[i0];
    float y1 = (v1 - mu) * inv * g[i1] + b[i1];
    xr[i0] = y0;  xsH[(size_t)t * DMODEL + i0] = (_Float16)y0;
    xr[i1] = y1;  xsH[(size_t)t * DMODEL + i1] = (_Float16)y1;
}

// ---------------------------------------------------------------------------
// Attention, one block per (batch, head). Seq = 65, dh = 64 -> tiny; VALU.
// Scores staged in LDS (65*65 f32 ~ 16.5 KB of the 320 KB WGP LDS).
// Output written as f16 directly (input to the out-projection GEMM).
// ---------------------------------------------------------------------------
__global__ void __launch_bounds__(256)
attention_kernel(const float* __restrict__ qkv, _Float16* __restrict__ oH) {
    __shared__ float sc[NP1 * NP1];
    int bh = blockIdx.x;
    int bb = bh >> 3;      // batch
    int hh = bh & 7;       // head
    const float* base = qkv + (size_t)bb * NP1 * (3 * DMODEL);

    for (int idx = threadIdx.x; idx < NP1 * NP1; idx += 256) {
        int i = idx / NP1, j = idx % NP1;
        const float* qr = base + (size_t)i * (3 * DMODEL) + hh * DHEAD;
        const float* kr = base + (size_t)j * (3 * DMODEL) + DMODEL + hh * DHEAD;
        float acc = 0.0f;
#pragma unroll 8
        for (int d = 0; d < DHEAD; d++) acc += qr[d] * kr[d];
        sc[idx] = acc * 0.125f;   // 1/sqrt(64)
    }
    __syncthreads();
    if (threadIdx.x < NP1) {
        int i = threadIdx.x;
        float mx = -1e30f;
        for (int j = 0; j < NP1; j++) mx = fmaxf(mx, sc[i * NP1 + j]);
        float s = 0.0f;
        for (int j = 0; j < NP1; j++) { float e = __expf(sc[i * NP1 + j] - mx); sc[i * NP1 + j] = e; s += e; }
        float inv = 1.0f / s;
        for (int j = 0; j < NP1; j++) sc[i * NP1 + j] *= inv;
    }
    __syncthreads();
    for (int idx = threadIdx.x; idx < NP1 * DHEAD; idx += 256) {
        int i = idx / DHEAD, d = idx % DHEAD;
        float acc = 0.0f;
        for (int j = 0; j < NP1; j++)
            acc += sc[i * NP1 + j] * base[(size_t)j * (3 * DMODEL) + 2 * DMODEL + hh * DHEAD + d];
        oH[((size_t)(bb * NP1 + i)) * DMODEL + hh * DHEAD + d] = (_Float16)acc;
    }
}

// ---------------------------------------------------------------------------
// Projection head: per (b, n<64) row, each thread i does the 256-wide softmax
// over j of h_raw*(x_i - x_j)/eps, gets expected-next / drift; block-reduce
// the drift correction; h_pot = h_raw - correction.
// ---------------------------------------------------------------------------
__global__ void __launch_bounds__(256)
hpot_kernel(const float* __restrict__ hfull, const float* __restrict__ marg,
            const float* __restrict__ xg, float* __restrict__ out) {
    __shared__ float xs_[MGRID];
    __shared__ float rWD[256], rMS[256];
    int blk = blockIdx.x;
    int bb = blk >> 6;     // batch
    int nn = blk & 63;     // n index (< 64)
    int i  = threadIdx.x;
    xs_[i] = xg[i];
    __syncthreads();

    const float* hrow = hfull + ((size_t)(bb * NP1 + nn)) * MGRID;
    float a  = hrow[i];
    float xi = xs_[i];
    float s  = a * (1.0f / EPS_PROJ);
    float mx = -1e30f;
    for (int j = 0; j < MGRID; j++) mx = fmaxf(mx, s * (xi - xs_[j]));
    float den = 0.0f, num = 0.0f;
    for (int j = 0; j < MGRID; j++) {
        float e = __expf(s * (xi - xs_[j]) - mx);
        den += e; num += e * xs_[j];
    }
    float drift = num / den - xi;
    float mu = marg[((size_t)bb * NP1 + nn) * MGRID + i];
    rWD[i] = mu * drift;
    rMS[i] = mu;
    __syncthreads();
    for (int sft = 128; sft > 0; sft >>= 1) {
        if (i < sft) { rWD[i] += rWD[i + sft]; rMS[i] += rMS[i + sft]; }
        __syncthreads();
    }
    float corr = rWD[0] / (rMS[0] + 1e-8f);
    out[(size_t)blk * MGRID + i] = a - corr;
}

// ---------------------------------------------------------------------------
// Host-side orchestration (all on `stream`; scratch carved from d_ws).
// Workspace requirement: ~41 MB.
// ---------------------------------------------------------------------------
extern "C" void kernel_launch(void* const* d_in, const int* in_sizes, int n_in,
                              void* d_out, int out_size, void* d_ws, size_t ws_size,
                              hipStream_t stream) {
    const float* marg   = (const float*)d_in[0];
    const float* xg     = (const float*)d_in[1];
    const float* conv_w = (const float*)d_in[2];
    const float* conv_b = (const float*)d_in[3];
    const float* fc_w   = (const float*)d_in[4];
    const float* fc_b   = (const float*)d_in[5];
    const float* ln0_g  = (const float*)d_in[6];
    const float* ln0_b  = (const float*)d_in[7];
    const float* pos    = (const float*)d_in[8];
    const float* qkv_w  = (const float*)d_in[9];
    const float* qkv_b  = (const float*)d_in[10];
    const float* out_w  = (const float*)d_in[11];
    const float* out_b  = (const float*)d_in[12];
    const float* ln1_g  = (const float*)d_in[13];
    const float* ln1_b  = (const float*)d_in[14];
    const float* ff1_w  = (const float*)d_in[15];
    const float* ff1_b  = (const float*)d_in[16];
    const float* ff2_w  = (const float*)d_in[17];
    const float* ff2_b  = (const float*)d_in[18];
    const float* ln2_g  = (const float*)d_in[19];
    const float* ln2_b  = (const float*)d_in[20];
    const float* u_w    = (const float*)d_in[21];
    const float* u_b    = (const float*)d_in[22];
    const float* h_w    = (const float*)d_in[23];
    const float* h_b    = (const float*)d_in[24];

    char* wp = (char*)d_ws;
    auto carve = [&](size_t bytes) -> char* {
        char* p = wp;
        wp += (bytes + 255) & ~(size_t)255;
        return p;
    };
    _Float16* qkvW  = (_Float16*)carve((size_t)NLAYER * 3 * DMODEL * DMODEL * 2);
    _Float16* outW  = (_Float16*)carve((size_t)NLAYER * DMODEL * DMODEL * 2);
    _Float16* ff1W  = (_Float16*)carve((size_t)NLAYER * DFFN * DMODEL * 2);
    _Float16* ff2W  = (_Float16*)carve((size_t)NLAYER * DMODEL * DFFN * 2);
    _Float16* uW    = (_Float16*)carve((size_t)MGRID * DMODEL * 2);
    _Float16* hW    = (_Float16*)carve((size_t)MGRID * DMODEL * 2);
    _Float16* fcWT  = (_Float16*)carve((size_t)DMODEL * 128 * 2);
    _Float16* embH  = (_Float16*)carve((size_t)TOK * 128 * 2);
    float*    preF  = (float*)   carve((size_t)TOK * DMODEL * 4);
    float*    xsF   = (float*)   carve((size_t)TOK * DMODEL * 4);
    _Float16* xsH   = (_Float16*)carve((size_t)TOK * DMODEL * 2);
    float*    qkvF  = (float*)   carve((size_t)TOK * 3 * DMODEL * 4);
    _Float16* oH    = (_Float16*)carve((size_t)TOK * DMODEL * 2);
    float*    projF = (float*)   carve((size_t)TOK * DMODEL * 4);
    _Float16* ff1H  = (_Float16*)carve((size_t)TOK * DFFN * 2);
    float*    hfull = (float*)   carve((size_t)TOK * MGRID * 4);
    (void)ws_size; (void)in_sizes; (void)n_in; (void)out_size;

    // ---- weight conversion to f16 ----
    auto cvt = [&](const float* s, _Float16* d, long n) {
        convert_f16_kernel<<<dim3((unsigned)((n + 255) / 256)), 256, 0, stream>>>(s, d, n);
    };
    cvt(qkv_w, qkvW, (long)NLAYER * 3 * DMODEL * DMODEL);
    cvt(out_w, outW, (long)NLAYER * DMODEL * DMODEL);
    cvt(ff1_w, ff1W, (long)NLAYER * DFFN * DMODEL);
    cvt(ff2_w, ff2W, (long)NLAYER * DMODEL * DFFN);
    cvt(u_w,   uW,   (long)MGRID * DMODEL);
    cvt(h_w,   hW,   (long)MGRID * DMODEL);
    transpose_f16_kernel<<<dim3((128 * DMODEL + 255) / 256), 256, 0, stream>>>(fc_w, fcWT, 128, DMODEL);

    // ---- embedder: conv+mean, then fc GEMM, then LN->GELU->(+pos) ----
    embed_kernel<<<dim3(TOK), 256, 0, stream>>>(marg, conv_w, conv_b, embH);
    dim3 blk(128);
    const int gy = (TOK + 63) / 64;  // 17
    gemm_wmma_f16<<<dim3(DMODEL / 64, gy), blk, 0, stream>>>(embH, fcWT, fc_b, preF, nullptr,
                                                             TOK, DMODEL, 128, 0);
    ln_gelu_pos_kernel<<<dim3(TOK), 256, 0, stream>>>(preF, ln0_g, ln0_b, pos, xsF, xsH);

    // ---- transformer layers ----
    for (int l = 0; l < NLAYER; l++) {
        gemm_wmma_f16<<<dim3(3 * DMODEL / 64, gy), blk, 0, stream>>>(
            xsH, qkvW + (size_t)l * 3 * DMODEL * DMODEL, qkv_b + (size_t)l * 3 * DMODEL,
            qkvF, nullptr, TOK, 3 * DMODEL, DMODEL, 0);
        attention_kernel<<<dim3(BATCH * NHEAD), 256, 0, stream>>>(qkvF, oH);
        gemm_wmma_f16<<<dim3(DMODEL / 64, gy), blk, 0, stream>>>(
            oH, outW + (size_t)l * DMODEL * DMODEL, out_b + (size_t)l * DMODEL,
            projF, nullptr, TOK, DMODEL, DMODEL, 0);
        add_ln_kernel<<<dim3(TOK), 256, 0, stream>>>(xsF, projF,
            ln1_g + (size_t)l * DMODEL, ln1_b + (size_t)l * DMODEL, xsH);
        gemm_wmma_f16<<<dim3(DFFN / 64, gy), blk, 0, stream>>>(
            xsH, ff1W + (size_t)l * DFFN * DMODEL, ff1_b + (size_t)l * DFFN,
            nullptr, ff1H, TOK, DFFN, DMODEL, 1 /*gelu*/);
        gemm_wmma_f16<<<dim3(DMODEL / 64, gy), blk, 0, stream>>>(
            ff1H, ff2W + (size_t)l * DMODEL * DFFN, ff2_b + (size_t)l * DMODEL,
            projF, nullptr, TOK, DMODEL, DFFN, 0);
        add_ln_kernel<<<dim3(TOK), 256, 0, stream>>>(xsF, projF,
            ln2_g + (size_t)l * DMODEL, ln2_b + (size_t)l * DMODEL, xsH);
    }

    // ---- heads ----
    float* u_out = (float*)d_out;                       // (B, NP1, M) = 266240 f32
    float* h_out = (float*)d_out + (size_t)TOK * MGRID; // (B, 64, M)
    gemm_wmma_f16<<<dim3(MGRID / 64, gy), blk, 0, stream>>>(xsH, uW, u_b, u_out, nullptr,
                                                            TOK, MGRID, DMODEL, 0);
    gemm_wmma_f16<<<dim3(MGRID / 64, gy), blk, 0, stream>>>(xsH, hW, h_b, hfull, nullptr,
                                                            TOK, MGRID, DMODEL, 0);
    hpot_kernel<<<dim3(BATCH * 64), 256, 0, stream>>>(hfull, marg, xg, h_out);
}
